// InfoNCELoss_68874095558996
// MI455X (gfx1250) — compile-verified, matrix operand order checked by reading, not run on hardware
//
#include <hip/hip_runtime.h>
#include <math.h>

// Problem constants (match the reference).
#define BATCH   4096
#define DIM     1024
#define NROWS   8192          // 2*BATCH
#define NTILES  512           // NROWS / 16
#define KTILES  32            // DIM / 32
#define INV_T   10.0f         // 1 / TEMPERATURE
#define NORM_EPS 1e-12f

typedef __attribute__((ext_vector_type(16))) _Float16 v16h;
typedef __attribute__((ext_vector_type(8)))  float    v8f;
typedef __attribute__((ext_vector_type(4)))  unsigned int v4u;

// ---------------------------------------------------------------------------
// Kernel 0: zero the scalar output accumulator (d_out is poisoned by harness).
// ---------------------------------------------------------------------------
__global__ void infonce_zero_out(float* out) {
    if (threadIdx.x == 0 && blockIdx.x == 0) out[0] = 0.0f;
}

// ---------------------------------------------------------------------------
// Kernel 1: L2-normalize each of the 8192 rows and store as f16 (row-major,
// stride DIM).  One 256-thread block (8 waves) per row.
// ---------------------------------------------------------------------------
__global__ void infonce_normalize_f16(const float* __restrict__ f1,
                                      const float* __restrict__ f2,
                                      _Float16* __restrict__ F) {
    const int row = blockIdx.x;                     // 0..8191
    const float* src = (row < BATCH) ? (f1 + (size_t)row * DIM)
                                     : (f2 + (size_t)(row - BATCH) * DIM);
    float ss = 0.0f;
    for (int i = threadIdx.x; i < DIM; i += 256) {
        float v = src[i];
        ss += v * v;
    }
    // wave32 xor-shuffle reduction
    #pragma unroll
    for (int off = 16; off > 0; off >>= 1) ss += __shfl_xor(ss, off, 32);

    __shared__ float red[8];
    const int wave = threadIdx.x >> 5;
    const int lane = threadIdx.x & 31;
    if (lane == 0) red[wave] = ss;
    __syncthreads();
    float tot = 0.0f;
    #pragma unroll
    for (int w = 0; w < 8; ++w) tot += red[w];

    const float scale = 1.0f / fmaxf(sqrtf(tot), NORM_EPS);
    _Float16* dst = F + ((size_t)row << 10);        // row * DIM
    for (int i = threadIdx.x; i < DIM; i += 256) {
        dst[i] = (_Float16)(src[i] * scale);
    }
}

// ---------------------------------------------------------------------------
// Kernel 2: fused sim-GEMM + logsumexp + positive extraction.
// One wave (32 threads) per block; each wave owns one 16-row tile of sim.
// The 16x1024 f16 A-strip (32 KB) is staged in LDS once; the B stream
// (L2-resident, 16 MB total) is read from global per column tile.
//
// Key numeric simplification: rows are unit-normalized, so sim/T = 10*dot is
// bounded by [-10, 10].  Use the fixed offset 10:
//     lse_i = 10 + log( sum_{j != i} exp(sim_ij/T - 10) )
// exp args are <= 0 -> no overflow, branchless accumulation.  The diagonal
// term is accumulated too and its exp subtracted once after the loop
// (diag exponent arg is recorded only by the owning lane; others hold -inf).
// ---------------------------------------------------------------------------
__global__ void __launch_bounds__(32)
infonce_lse(const _Float16* __restrict__ F, float* __restrict__ out) {
    __shared__ v4u As[2048];                        // 16 rows * 1024 halves

    const int lane  = threadIdx.x & 31;
    const int itile = blockIdx.x;                   // 0..511
    const int i0    = itile << 4;

    // Stage the A strip (rows i0..i0+15, contiguous 32 KB) into LDS.
    const v4u* srcA = (const v4u*)(F + ((size_t)i0 << 10));
    for (int idx = lane; idx < 2048; idx += 32) As[idx] = srcA[idx];
    __syncthreads();

    // C-fragment geometry: VGPR r, lanes 0-15 -> (M=r, N=lane),
    //                               lanes 16-31 -> (M=r+8, N=lane-16).
    const int n_col  = lane & 15;                   // N within the tile
    const int m_half = (lane >> 4) << 3;            // 0 or 8 row offset
    const int khalf  = (lane >> 4) & 1;             // K-half select for frags
    const int ptile  = (itile + (BATCH >> 4)) & (NTILES - 1);  // partner tile

    bool eq[8];                                     // slot sits on M==N line?
    #pragma unroll
    for (int r = 0; r < 8; ++r) eq[r] = (n_col == m_half + r);

    float sm[8], posv[8], diagv[8];
    #pragma unroll
    for (int r = 0; r < 8; ++r) { sm[r] = 0.0f; posv[r] = 0.0f; diagv[r] = -INFINITY; }

    for (int jt = 0; jt < NTILES; ++jt) {
        const int j0 = jt << 4;

        // Prefetch next column tile's row (stays warm in WGP$/L2).
        if (jt + 1 < NTILES) {
            __builtin_prefetch(F + (((size_t)(j0 + 16 + n_col)) << 10), 0, 1);
        }

        v8f c = {};
        #pragma unroll 8
        for (int kt = 0; kt < KTILES; ++kt) {
            // A fragment from LDS (16-bit 16x32 operand layout, ISA 7.12.2).
            union { v4u u[2]; v16h h; } xa, xb;
            const int ab = ((lane & 15) << 7) + (kt << 2) + khalf;
            xa.u[0] = As[ab];
            xa.u[1] = As[ab + 2];

            // B fragment from global: row j0+(lane&15), symmetric packing.
            const int kbase = (kt << 5) + (khalf << 3);
            const _Float16* pb = F + (((size_t)(j0 + n_col)) << 10) + kbase;
            xb.u[0] = *(const v4u*)(pb);
            xb.u[1] = *(const v4u*)(pb + 16);

            c = __builtin_amdgcn_wmma_f32_16x16x32_f16(
                    /*neg_a=*/false, xa.h, /*neg_b=*/false, xb.h,
                    /*c_mod=*/(short)0, c, /*reuse_a=*/false, /*reuse_b=*/false);
        }

        // Branchless accumulation: exp(sim/T - 10) = exp(10*c - 10).
        #pragma unroll
        for (int r = 0; r < 8; ++r) {
            sm[r] += __expf(fmaf(c[r], INV_T, -INV_T));
        }
        if (jt == itile) {                          // record diagonal exp-arg
            #pragma unroll
            for (int r = 0; r < 8; ++r)
                if (eq[r]) diagv[r] = fmaf(c[r], INV_T, -INV_T);
        }
        if (jt == ptile) {                          // record positive sim/T
            #pragma unroll
            for (int r = 0; r < 8; ++r)
                if (eq[r]) posv[r] = c[r] * INV_T;
        }
    }

    // Remove the diagonal contribution (exp(-inf) = 0 for non-owning lanes).
    #pragma unroll
    for (int r = 0; r < 8; ++r) sm[r] -= __expf(diagv[r]);

    // Merge the 16 per-lane partial sums of each row.  xor masks 1,2,4,8 keep
    // lanes 0-15 (rows M=r) and 16-31 (rows M=r+8) independent.
    float partial = 0.0f;
    #pragma unroll
    for (int r = 0; r < 8; ++r) {
        float s = sm[r];
        float p = posv[r];
        #pragma unroll
        for (int off = 1; off < 16; off <<= 1) {
            s += __shfl_xor(s, off, 32);
            p += __shfl_xor(p, off, 32);
        }
        if (n_col == 0) {                           // lanes 0 and 16
            partial += (INV_T + __logf(s)) - p;     // lse - pos
        }
    }
    if (n_col == 0) {
        atomicAdd(out, partial * (1.0f / (float)NROWS));  // mean reduction
    }
}

// ---------------------------------------------------------------------------
// Host-side launcher.
// ---------------------------------------------------------------------------
extern "C" void kernel_launch(void* const* d_in, const int* in_sizes, int n_in,
                              void* d_out, int out_size, void* d_ws, size_t ws_size,
                              hipStream_t stream) {
    (void)in_sizes; (void)n_in; (void)out_size; (void)ws_size;
    const float* f1 = (const float*)d_in[0];
    const float* f2 = (const float*)d_in[1];
    float* out = (float*)d_out;
    _Float16* F = (_Float16*)d_ws;                  // 8192*1024*2B = 16 MB

    infonce_zero_out<<<1, 64, 0, stream>>>(out);
    infonce_normalize_f16<<<NROWS, 256, 0, stream>>>(f1, f2, F);
    infonce_lse<<<NTILES, 32, 0, stream>>>(F, out);
}
